// MultiHeadAttention_47579647705431
// MI455X (gfx1250) — matrix operand order repaired and checked by reading, not compiled, not run on hardware
//
#include <hip/hip_runtime.h>
#include <hip/hip_bf16.h>

// MI455X (gfx1250) flash-attention MHA.
//  - all matmuls via v_wmma_f32_16x16x32_f16 (wave32, one 16x16 tile / wave)
//  - f16 intermediates in workspace (QP/KP head-split, V transposed, concat)
//  - streaming softmax (no BxHxSxS materialization), stats in registers
//  - causal mask analytic; zero_pad zeroes query-row-0 attention output

typedef __attribute__((ext_vector_type(16))) _Float16 v16h;
typedef __attribute__((ext_vector_type(8)))  float    v8f;

#define BB 16
#define SS 1024
#define DM 512
#define NH 8
#define DH 64

static __device__ __forceinline__ v8f wmma_f16(v16h a, v16h b, v8f c) {
  // D = A(16x32) * B(32x16) + C, fp32 accumulate
  return __builtin_amdgcn_wmma_f32_16x16x32_f16(false, a, false, b,
                                                (short)0, c, false, false);
}

// ---------------------------------------------------------------------------
// Projection: out = X @ W^T + bias, X fp32 [16384 x 512], W fp32 [512 x 512].
// vmode==0: write head-split f16 [B,H,S,Dh]; vmode==1: transposed [B,H,Dh,S].
// One wave per 16x16 output tile; K chained in 32-steps.
// ---------------------------------------------------------------------------
__global__ __launch_bounds__(256) void proj_xwT_kernel(
    const float* __restrict__ X, const float* __restrict__ W,
    const float* __restrict__ bias, _Float16* __restrict__ out, int vmode) {
  const int lane = threadIdx.x;
  const int r    = lane & 15;
  const int hi   = lane >> 4;
  const int base = hi * 8;    // unused; A-fragment also uses kb2
  const int kb2  = hi * 16;   // B-fragment K base
  const int m_tile = blockIdx.x;
  const int n_tile = blockIdx.y * 8 + threadIdx.y;
  const int ncol   = n_tile * 16 + r;

  const float* xr = X + (size_t)(m_tile * 16 + r) * DM;
  const float* wr = W + (size_t)ncol * DM + kb2;

  v8f c = {};
#pragma unroll 4
  for (int kk = 0; kk < DM; kk += 32) {
    v16h a, bf;
#pragma unroll
    for (int i = 0; i < 8; ++i) {
      a[i]     = (_Float16)xr[kk + kb2 + i];
      a[8 + i] = (_Float16)xr[kk + kb2 + 8 + i];
    }
#pragma unroll
    for (int i = 0; i < 16; ++i) bf[i] = (_Float16)wr[kk + i];
    c = wmma_f16(a, bf, c);
  }

  const float bval = bias[ncol];
  const int h = ncol >> 6;
  const int d = ncol & 63;
#pragma unroll
  for (int j = 0; j < 8; ++j) {
    const int m    = m_tile * 16 + j + hi * 8;   // global row (b*S + s)
    const int bidx = m >> 10;
    const int s    = m & (SS - 1);
    const float v  = c[j] + bval;
    size_t idx;
    if (vmode) idx = ((size_t)(bidx * NH + h) * DH + d) * SS + s;
    else       idx = ((size_t)(bidx * NH + h) * SS + s) * DH + d;
    out[idx] = (_Float16)v;
  }
}

// ---------------------------------------------------------------------------
// Flash attention: one wave handles 16 query rows of one (b,h); streams keys
// 32 at a time. Scores/softmax in fp32 registers, P re-staged through a
// wave-private LDS tile to become the next WMMA A-fragment.
// ---------------------------------------------------------------------------
__global__ __launch_bounds__(128) void flash_attn_kernel(
    const _Float16* __restrict__ QP, const _Float16* __restrict__ KP,
    const _Float16* __restrict__ VPT, _Float16* __restrict__ CT,
    const int* __restrict__ zero_pad) {
  __shared__ _Float16 plds[4][16 * 32];   // 1 KB per wave
  const int lane = threadIdx.x;
  const int r    = lane & 15;
  const int hi   = lane >> 4;
  const int base = hi * 8;
  const int kb2  = hi * 16;
  const int h    = blockIdx.y;
  const int b    = blockIdx.z;
  const int qbase = (blockIdx.x * 4 + threadIdx.y) * 16;

  const _Float16* Qh = QP  + (size_t)(b * NH + h) * SS * DH;
  const _Float16* Kh = KP  + (size_t)(b * NH + h) * SS * DH;
  const _Float16* Vh = VPT + (size_t)(b * NH + h) * DH * SS;   // [d][s]
  _Float16* pl = plds[threadIdx.y];

  // Q as A-fragments (Dh=64 -> two K=32 fragments), pre-scaled by 1/sqrt(Dh)
  const _Float16* qr = Qh + (size_t)(qbase + r) * DH;
  v16h aq0, aq1;
#pragma unroll
  for (int i = 0; i < 8; ++i) {
    aq0[i]     = (_Float16)((float)qr[kb2 + i] * 0.125f);
    aq0[8 + i] = (_Float16)((float)qr[kb2 + 8 + i] * 0.125f);
    aq1[i]     = (_Float16)((float)qr[32 + kb2 + i] * 0.125f);
    aq1[8 + i] = (_Float16)((float)qr[40 + kb2 + i] * 0.125f);
  }

  v8f o0 = {}, o1 = {}, o2 = {}, o3 = {};
  float mrow[8], lrow[8];
#pragma unroll
  for (int j = 0; j < 8; ++j) { mrow[j] = -3.0e38f; lrow[j] = 0.0f; }

  const int kend = qbase + 15;               // causal: keys <= last query row
  for (int kb = 0; kb <= kend; kb += 32) {
    if (kb + 32 <= kend) {                   // prefetch next K/V tiles
      __builtin_prefetch(Kh + (size_t)(kb + 32 + r) * DH, 0, 1);
      __builtin_prefetch(Vh + (size_t)r * SS + kb + 32, 0, 1);
    }
    // K rows as B-fragments (contiguous in d)
    const _Float16* kr0 = Kh + (size_t)(kb + r) * DH;
    const _Float16* kr1 = Kh + (size_t)(kb + 16 + r) * DH;
    v16h bk00 = *(const v16h*)(kr0 + kb2);
    v16h bk01 = *(const v16h*)(kr0 + 32 + kb2);
    v16h bk10 = *(const v16h*)(kr1 + kb2);
    v16h bk11 = *(const v16h*)(kr1 + 32 + kb2);

    v8f s0 = {}, s1 = {};
    s0 = wmma_f16(aq0, bk00, s0);
    s0 = wmma_f16(aq1, bk01, s0);
    s1 = wmma_f16(aq0, bk10, s1);
    s1 = wmma_f16(aq1, bk11, s1);

    float alpha[8];
#pragma unroll
    for (int j = 0; j < 8; ++j) {
      const int row_g = qbase + j + hi * 8;
      if (kb + r > row_g)      s0[j] = -1.0e30f;   // causal mask
      if (kb + 16 + r > row_g) s1[j] = -1.0e30f;
      float rmax = fmaxf(s0[j], s1[j]);
#pragma unroll
      for (int off = 8; off >= 1; off >>= 1)
        rmax = fmaxf(rmax, __shfl_xor(rmax, off, 16));
      const float mnew = fmaxf(mrow[j], rmax);
      const float p0 = __expf(s0[j] - mnew);
      const float p1 = __expf(s1[j] - mnew);
      float rsum = p0 + p1;
#pragma unroll
      for (int off = 8; off >= 1; off >>= 1)
        rsum += __shfl_xor(rsum, off, 16);
      const float a = __expf(mrow[j] - mnew);
      lrow[j] = lrow[j] * a + rsum;
      mrow[j] = mnew;
      alpha[j] = a;
      // stage P (16x32, row=query, col=key) for A-fragment regather
      pl[(j + hi * 8) * 32 + r]      = (_Float16)p0;
      pl[(j + hi * 8) * 32 + 16 + r] = (_Float16)p1;
    }

    // regather P as WMMA A-fragment (wave-private LDS, in-order DS ops)
    v16h pa;
#pragma unroll
    for (int i = 0; i < 8; ++i) {
      pa[i]     = pl[r * 32 + kb2 + i];
      pa[8 + i] = pl[r * 32 + kb2 + 8 + i];
    }

    // V^T rows as B-fragments (contiguous in key dim)
    const _Float16* vb = Vh + kb + kb2;
    v16h bv0 = *(const v16h*)(vb + (size_t)(r) * SS);
    v16h bv1 = *(const v16h*)(vb + (size_t)(16 + r) * SS);
    v16h bv2 = *(const v16h*)(vb + (size_t)(32 + r) * SS);
    v16h bv3 = *(const v16h*)(vb + (size_t)(48 + r) * SS);

#pragma unroll
    for (int j = 0; j < 8; ++j) {   // online-softmax rescale of accumulators
      o0[j] *= alpha[j]; o1[j] *= alpha[j];
      o2[j] *= alpha[j]; o3[j] *= alpha[j];
    }
    o0 = wmma_f16(pa, bv0, o0);
    o1 = wmma_f16(pa, bv1, o1);
    o2 = wmma_f16(pa, bv2, o2);
    o3 = wmma_f16(pa, bv3, o3);
  }

  const int zp = zero_pad[0];
#pragma unroll
  for (int j = 0; j < 8; ++j) {
    const int row_g = qbase + j + hi * 8;
    const float inv = (zp && row_g == 0) ? 0.0f : (1.0f / lrow[j]);
    _Float16* cr = CT + (size_t)(b * SS + row_g) * DM + h * DH;
    cr[ 0 + r] = (_Float16)(o0[j] * inv);
    cr[16 + r] = (_Float16)(o1[j] * inv);
    cr[32 + r] = (_Float16)(o2[j] * inv);
    cr[48 + r] = (_Float16)(o3[j] * inv);
  }
}

// ---------------------------------------------------------------------------
// Output projection: out(f32) = CT(f16) @ Wo^T + bo
// ---------------------------------------------------------------------------
__global__ __launch_bounds__(256) void outproj_kernel(
    const _Float16* __restrict__ A, const float* __restrict__ W,
    const float* __restrict__ bias, float* __restrict__ out) {
  const int lane = threadIdx.x;
  const int r    = lane & 15;
  const int hi   = lane >> 4;
  const int base = hi * 8;
  const int kb2  = hi * 16;
  const int m_tile = blockIdx.x;
  const int n_tile = blockIdx.y * 8 + threadIdx.y;
  const int ncol   = n_tile * 16 + r;

  const _Float16* ar = A + (size_t)(m_tile * 16 + r) * DM;
  const float*    wr = W + (size_t)ncol * DM + kb2;

  v8f c = {};
#pragma unroll 4
  for (int kk = 0; kk < DM; kk += 32) {
    v16h a, bf;
#pragma unroll
    for (int i = 0; i < 8; ++i) {
      a[i]     = ar[kk + kb2 + i];
      a[8 + i] = ar[kk + kb2 + 8 + i];
    }
#pragma unroll
    for (int i = 0; i < 16; ++i) bf[i] = (_Float16)wr[kk + i];
    c = wmma_f16(a, bf, c);
  }
  const float bval = bias[ncol];
#pragma unroll
  for (int j = 0; j < 8; ++j)
    out[(size_t)(m_tile * 16 + j + hi * 8) * DM + ncol] = c[j] + bval;
}

// ---------------------------------------------------------------------------
extern "C" void kernel_launch(void* const* d_in, const int* in_sizes, int n_in,
                              void* d_out, int out_size, void* d_ws, size_t ws_size,
                              hipStream_t stream) {
  const float* q  = (const float*)d_in[0];
  const float* k  = (const float*)d_in[1];
  const float* v  = (const float*)d_in[2];
  // d_in[3] = mask (causal, derived analytically; unused)
  const int* zero_pad = (const int*)d_in[4];
  const float* Wk = (const float*)d_in[5];
  const float* bk = (const float*)d_in[6];
  const float* Wv = (const float*)d_in[7];
  const float* bv = (const float*)d_in[8];
  const float* Wo = (const float*)d_in[9];
  const float* bo = (const float*)d_in[10];
  float* out = (float*)d_out;

  // Workspace layout (f16): QP | KP | VPT | CT, each B*S*DM = 8388608 elems.
  const size_t N = (size_t)BB * SS * DM;       // 8,388,608
  _Float16* QP  = (_Float16*)d_ws;
  _Float16* KP  = QP + N;
  _Float16* VPT = KP + N;
  _Float16* CT  = VPT + N;                     // total 64 MB of d_ws

  const dim3 gblk(32, 8);                      // 8 waves / block
  const dim3 ggrd(DM * BB * SS / DM / 16, DM / 16 / 8);  // (1024, 4)

  proj_xwT_kernel<<<ggrd, gblk, 0, stream>>>(q, Wk, bk, QP, 0);
  proj_xwT_kernel<<<ggrd, gblk, 0, stream>>>(k, Wk, bk, KP, 0);
  proj_xwT_kernel<<<ggrd, gblk, 0, stream>>>(v, Wv, bv, VPT, 1);

  flash_attn_kernel<<<dim3(SS / 64, NH, BB), dim3(32, 4), 0, stream>>>(
      QP, KP, VPT, CT, zero_pad);

  outproj_kernel<<<ggrd, gblk, 0, stream>>>(CT, Wo, bo, out);
}
